// MyMultiHeadAttention_48103633715879
// MI455X (gfx1250) — matrix operand order repaired and checked by reading, compiled, not run on hardware
//
#include <hip/hip_runtime.h>

// ---------------------------------------------------------------------------
// Multi-head attention, CDNA5 (gfx1250), wave32, WMMA f32_16x16x32_f16.
// B=4 S=2048 D=256 H=8 E=64.  Compute-bound -> all GEMMs through v_wmma.
// All WMMA operands are pre-packed in fragment-linear order (lane ln owns
// f16[ln*16 .. ln*16+15] of a 512-element tile), so every fragment load is a
// single 32-byte v16h load. Attention K/V chunks are double-buffered in LDS
// via global_load_async_to_lds_b128 (ASYNCcnt) and shared by all 4 waves.
// ---------------------------------------------------------------------------

#define Bn 4
#define Sn 2048
#define Dn 256
#define Hn 8
#define En 64
#define HE 512

typedef __attribute__((ext_vector_type(16))) _Float16 v16h;
typedef __attribute__((ext_vector_type(4)))  _Float16 v4h;
typedef __attribute__((ext_vector_type(8)))  float    v8f;

// A-matrix 16x32 f16 (ISA 7.12.2): lane half hi, element j -> K index.
__device__ __forceinline__ int kkmap(int j, int hi) {
    return (j < 8 ? j : j + 8) + (hi << 3);
}
// Inverse A map: (row 0..15, K 0..31) -> flat index in 512-elem A-frag tile.
__device__ __forceinline__ int ainv(int row, int c32) {
    int hi = c32 >> 4;
    int j  = c32 & 15;
    return (((hi << 4) + row) << 4) + j;
}
// Inverse B map: (K 0..31, col 0..15) -> flat index in 512-elem B-frag tile.
__device__ __forceinline__ int binv(int k32, int n) {
    return ((((k32 >> 4) << 4) + n) << 4) + (k32 & 15);
}

__device__ __forceinline__ v8f wmma16(v16h a, v16h b, v8f c) {
    return __builtin_amdgcn_wmma_f32_16x16x32_f16(false, a, false, b,
                                                  (short)0, c, false, false);
}

// ---------------------------------------------------------------------------
// Kernel 0: pack W [H,3,D,E] and Wo [512,64] into B-fragment-linear f16 tiles.
// Wpk tile id = ((h*3+mat)*8 + ks)*4 + nt ; Wopk tile id = ks*4 + nt.
// ---------------------------------------------------------------------------
__global__ __launch_bounds__(256) void mha_convert(
    const float* __restrict__ W, const float* __restrict__ Wo,
    _Float16* __restrict__ Wpk, _Float16* __restrict__ Woh) {
    int o = blockIdx.x * 256 + threadIdx.x;
    if (o < Hn * 3 * Dn * En) {
        int j = o & 15, ln = (o >> 4) & 31, tile = o >> 9;
        int nt = tile & 3, ks = (tile >> 2) & 7, t2 = tile >> 5;  // t2 = h*3+mat
        int k = ((ln >> 4) << 4) + j, n = ln & 15;
        Wpk[o] = (_Float16)W[((size_t)t2 * Dn + ks * 32 + k) * En + nt * 16 + n];
    }
    if (o < HE * En) {
        int j = o & 15, ln = (o >> 4) & 31, tile = o >> 9;
        int nt = tile & 3, ks = tile >> 2;
        int k = ((ln >> 4) << 4) + j, n = ln & 15;
        Woh[o] = (_Float16)Wo[(ks * 32 + k) * En + nt * 16 + n];
    }
}

// ---------------------------------------------------------------------------
// Kernel 1: QKV projection. Block = (b,h) x 64 S-rows; 4 waves, one 16-row M
// tile each. x tile is scattered into LDS directly in A-fragment order, so A
// loads are contiguous ds_load_b128 pairs; W fragments are contiguous global
// v16h loads. Epilogue scatters results into attention-ready packed layouts:
//   qpk (A-layout, pre-scaled 1/8), kpk (B-layout over e), vpk (B-layout).
// ---------------------------------------------------------------------------
__global__ __launch_bounds__(128) void mha_qkv(
    const float* __restrict__ x, const _Float16* __restrict__ Wpk,
    _Float16* __restrict__ qpk, _Float16* __restrict__ kpk,
    _Float16* __restrict__ vpk) {
    __shared__ __align__(32) _Float16 xs[4 * 8 * 512];  // [mtile][ks][frag512]

    const int tid = threadIdx.x;
    const int bh  = blockIdx.y;
    const int h   = bh & 7;
    const int b   = bh >> 3;
    const int s0  = blockIdx.x * 64;

    // Coalesced x load (float4) -> A-fragment-ordered LDS scatter (v4h).
    const float4* xv = (const float4*)(x + ((size_t)b * Sn + s0) * Dn);
    for (int i = tid; i < 64 * 64; i += 128) {
        int r = i >> 6, c4 = i & 63;
        float4 tv = xv[r * 64 + c4];
        int mt = r >> 4, rIn = r & 15;
        int c = c4 << 2, ks = c >> 5, c32 = c & 31;
        int fi = ainv(rIn, c32);               // 4 consecutive slots, 8B aligned
        v4h pk = {(_Float16)tv.x, (_Float16)tv.y, (_Float16)tv.z, (_Float16)tv.w};
        *(v4h*)&xs[(mt * 8 + ks) * 512 + fi] = pk;
    }
    __syncthreads();

    const int ln = tid & 31, wv = tid >> 5;
    const int l = ln & 15, hi = ln >> 4;
    const int kt16 = bh * (Sn / 16);

#pragma unroll
    for (int mat = 0; mat < 3; ++mat) {
        v8f acc[4] = {};
        const _Float16* wb = Wpk + (size_t)((h * 3 + mat) * 8) * 4 * 512;
        for (int ks = 0; ks < 8; ++ks) {       // K = 256, steps of 32
            v16h a = *(const v16h*)&xs[(wv * 8 + ks) * 512 + ln * 16];
#pragma unroll
            for (int nt = 0; nt < 4; ++nt) {
                v16h bw = *(const v16h*)(wb + (size_t)(ks * 4 + nt) * 512 + ln * 16);
                acc[nt] = wmma16(a, bw, acc[nt]);
            }
        }
#pragma unroll
        for (int nt = 0; nt < 4; ++nt)
#pragma unroll
            for (int r = 0; r < 8; ++r) {
                int row = s0 + wv * 16 + hi * 8 + r;   // s index
                int col = nt * 16 + l;                 // e index 0..63
                float val = acc[nt][r];
                if (mat == 0) {
                    int idx = ((kt16 + (row >> 4)) * 2 + (col >> 5)) * 512 +
                              ainv(row & 15, col & 31);
                    qpk[idx] = (_Float16)(val * 0.125f);
                } else if (mat == 1) {
                    int idx = ((kt16 + (row >> 4)) * 2 + (col >> 5)) * 512 +
                              binv(col & 31, row & 15);
                    kpk[idx] = (_Float16)val;
                } else {
                    int idx = ((bh * (Sn / 32) + (row >> 5)) * 4 + (col >> 4)) * 512 +
                              binv(row & 31, col & 15);
                    vpk[idx] = (_Float16)val;
                }
            }
    }
}

// ---------------------------------------------------------------------------
// Kernel 2: flash attention. One wave = 16 query rows, keys streamed in
// chunks of 32 with online softmax. K/V chunks (8 KB, contiguous in the
// packed layouts) are double-buffered in LDS with async global->LDS copies
// shared by all 4 waves; fragments then come from ds_load_b128. P goes to LDS
// in A-fragment order and returns as one v16h. Output -> opk (A-layout).
// ---------------------------------------------------------------------------
__global__ __launch_bounds__(128) void mha_attn(
    const _Float16* __restrict__ qpk, const _Float16* __restrict__ kpk,
    const _Float16* __restrict__ vpk, _Float16* __restrict__ opk) {
    __shared__ __align__(32) _Float16 ps[4][512];
    __shared__ __align__(32) _Float16 kv[2][4096];   // [buf][ K:2048 | V:2048 ]

    const int tid = threadIdx.x;
    const int ln = tid & 31, wv = tid >> 5;
    const int l = ln & 15, hi = ln >> 4;
    const int bh = blockIdx.y;
    const int b = bh >> 3, h = bh & 7;
    const int s0 = blockIdx.x * 64 + wv * 16;

    const _Float16* qb = qpk + (size_t)(bh * (Sn / 16) + (s0 >> 4)) * 2 * 512;
    const _Float16* kg = kpk + (size_t)bh * (Sn / 16) * 2 * 512;
    const _Float16* vg = vpk + (size_t)bh * (Sn / 32) * 4 * 512;

    v16h qa[2];
    qa[0] = *(const v16h*)(qb + ln * 16);
    qa[1] = *(const v16h*)(qb + 512 + ln * 16);

    float mrow[8], lrow[8];
#pragma unroll
    for (int r = 0; r < 8; ++r) { mrow[r] = -3.0e38f; lrow[r] = 0.0f; }
    v8f oacc[4] = {};
    _Float16* pw = &ps[wv][0];

    // Async-copy one 32-key K/V chunk (8 KB) into kv[buf]; 16B/lane/issue,
    // 4 issues per wave across the 4 waves of the block.
    auto issue_chunk = [&](int t, int buf) {
        const char* ksrc = (const char*)(kg + (size_t)(t >> 4) * 2 * 512);
        const char* vsrc = (const char*)(vg + (size_t)(t >> 5) * 4 * 512);
        unsigned lbase = (unsigned)(uintptr_t)&kv[buf][0];
#pragma unroll
        for (int it = 0; it < 4; ++it) {
            int o = it * 2048 + tid * 16;
            const char* src = (it < 2) ? (ksrc + o) : (vsrc + (o - 4096));
            unsigned la = lbase + o;
            unsigned long long ga = (unsigned long long)(uintptr_t)src;
            asm volatile("global_load_async_to_lds_b128 %0, %1, off"
                         :: "v"(la), "v"(ga) : "memory");
        }
    };

    issue_chunk(0, 0);
    asm volatile("s_wait_asynccnt 0x0" ::: "memory");
    __syncthreads();

    for (int t = 0; t < Sn; t += 32) {
        const int buf = (t >> 5) & 1;
        if (t + 32 < Sn) issue_chunk(t + 32, buf ^ 1);   // overlap with compute
        const _Float16* kc = &kv[buf][0];
        const _Float16* vc = &kv[buf][2048];

        v8f e[2] = {};
#pragma unroll
        for (int nt = 0; nt < 2; ++nt) {
#pragma unroll
            for (int kk = 0; kk < 2; ++kk) {
                v16h bk = *(const v16h*)(kc + (nt * 2 + kk) * 512 + ln * 16);
                e[nt] = wmma16(qa[kk], bk, e[nt]);
            }
        }

        // Online softmax (rows r+hi*8; 16 cols across the 16-lane half).
        float p0[8], p1[8], sc[8];
#pragma unroll
        for (int r = 0; r < 8; ++r) {
            float mx = fmaxf(e[0][r], e[1][r]);
            mx = fmaxf(mx, __shfl_xor(mx, 1, 32));
            mx = fmaxf(mx, __shfl_xor(mx, 2, 32));
            mx = fmaxf(mx, __shfl_xor(mx, 4, 32));
            mx = fmaxf(mx, __shfl_xor(mx, 8, 32));
            float mn = fmaxf(mrow[r], mx);
            sc[r] = __expf(mrow[r] - mn);
            mrow[r] = mn;
            p0[r] = __expf(e[0][r] - mn);
            p1[r] = __expf(e[1][r] - mn);
            float rs = p0[r] + p1[r];
            rs += __shfl_xor(rs, 1, 32);
            rs += __shfl_xor(rs, 2, 32);
            rs += __shfl_xor(rs, 4, 32);
            rs += __shfl_xor(rs, 8, 32);
            lrow[r] = lrow[r] * sc[r] + rs;
        }
#pragma unroll
        for (int nt = 0; nt < 4; ++nt)
#pragma unroll
            for (int r = 0; r < 8; ++r) oacc[nt][r] *= sc[r];

        // P: C layout -> LDS in A-fragment order -> one v16h back.
#pragma unroll
        for (int r = 0; r < 8; ++r) {
            pw[ainv(hi * 8 + r, l)]      = (_Float16)p0[r];
            pw[ainv(hi * 8 + r, 16 + l)] = (_Float16)p1[r];
        }
        asm volatile("s_wait_dscnt 0" ::: "memory");
        v16h pa = *(const v16h*)&pw[ln * 16];

#pragma unroll
        for (int nt = 0; nt < 4; ++nt) {
            v16h bv = *(const v16h*)(vc + nt * 512 + ln * 16);
            oacc[nt] = wmma16(pa, bv, oacc[nt]);
        }

        // Next chunk's async copy must land before anyone reads it.
        asm volatile("s_wait_asynccnt 0x0" ::: "memory");
        __syncthreads();
    }

    // Normalize; scatter into opk in A-fragment order (concat-heads columns).
    float rinv[8];
#pragma unroll
    for (int r = 0; r < 8; ++r) rinv[r] = 1.0f / lrow[r];
#pragma unroll
    for (int nt = 0; nt < 4; ++nt)
#pragma unroll
        for (int r = 0; r < 8; ++r) {
            int flat = b * Sn + s0 + hi * 8 + r;       // row in [0, B*S)
            int col  = h * En + nt * 16 + l;           // 0..511
            int idx  = ((flat >> 4) * 16 + (col >> 5)) * 512 +
                       ainv(flat & 15, col & 31);
            opk[idx] = (_Float16)(oacc[nt][r] * rinv[r]);
        }
}

// ---------------------------------------------------------------------------
// Kernel 3: output projection z = concat @ Wo, [8192 x 512] @ [512 x 64].
// A and B fragments are single v16h loads from packed layouts.
// ---------------------------------------------------------------------------
__global__ __launch_bounds__(128) void mha_proj(
    const _Float16* __restrict__ opk, const _Float16* __restrict__ Woh,
    float* __restrict__ z) {
    const int tid = threadIdx.x;
    const int ln = tid & 31, wv = tid >> 5;
    const int l = ln & 15, hi = ln >> 4;
    const int row0 = blockIdx.x * 64 + wv * 16;        // one A tile per wave

    const _Float16* ob = opk + (size_t)(row0 >> 4) * 16 * 512;
    v8f acc[4] = {};
    for (int ks = 0; ks < 16; ++ks) {                  // K = 512, steps of 32
        v16h a = *(const v16h*)(ob + (size_t)ks * 512 + ln * 16);
#pragma unroll
        for (int nt = 0; nt < 4; ++nt) {
            v16h bw = *(const v16h*)(Woh + (size_t)(ks * 4 + nt) * 512 + ln * 16);
            acc[nt] = wmma16(a, bw, acc[nt]);
        }
    }
#pragma unroll
    for (int nt = 0; nt < 4; ++nt)
#pragma unroll
        for (int r = 0; r < 8; ++r)
            z[(size_t)(row0 + hi * 8 + r) * En + nt * 16 + l] = acc[nt][r];
}

// ---------------------------------------------------------------------------
extern "C" void kernel_launch(void* const* d_in, const int* in_sizes, int n_in,
                              void* d_out, int out_size, void* d_ws, size_t ws_size,
                              hipStream_t stream) {
    const float* x  = (const float*)d_in[0];   // [B,S,D]
    const float* W  = (const float*)d_in[1];   // [H,3,D,E]
    const float* Wo = (const float*)d_in[2];   // [H*E,E]
    float* z = (float*)d_out;                  // [B,S,E]

    char* ws = (char*)d_ws;
    const size_t qkvBytes = (size_t)Bn * Hn * Sn * En * sizeof(_Float16); // 8 MiB
    _Float16* qpk = (_Float16*)(ws + 0 * qkvBytes);
    _Float16* kpk = (_Float16*)(ws + 1 * qkvBytes);
    _Float16* vpk = (_Float16*)(ws + 2 * qkvBytes);
    _Float16* opk = (_Float16*)(ws + 3 * qkvBytes);
    _Float16* Wpk = (_Float16*)(ws + 4 * qkvBytes);
    _Float16* Woh = (_Float16*)(ws + 4 * qkvBytes +
                                (size_t)Hn * 3 * Dn * En * sizeof(_Float16));

    mha_convert<<<dim3((Hn * 3 * Dn * En + 255) / 256), dim3(256), 0, stream>>>(
        W, Wo, Wpk, Woh);
    mha_qkv<<<dim3(Sn / 64, Bn * Hn), dim3(128), 0, stream>>>(
        x, Wpk, qpk, kpk, vpk);
    mha_attn<<<dim3(Sn / 64, Bn * Hn), dim3(128), 0, stream>>>(
        qpk, kpk, vpk, opk);
    mha_proj<<<dim3(Bn * Sn / 64), dim3(128), 0, stream>>>(
        opk, Woh, z);
}
